// NeuralDNALayer_27676769255881
// MI455X (gfx1250) — matrix operand order repaired and checked by reading, not compiled, not run on hardware
//
#include <hip/hip_runtime.h>

// ---------------- problem dimensions (fixed by the reference) ----------------
#define BT   8192   // B*T tokens
#define DD   2048   // model dim D
#define GG   4096   // gene bank size G
#define GD   256    // gene dim
#define KACT 32     // top-k

typedef __attribute__((ext_vector_type(16))) __bf16 v16bf;
typedef __attribute__((ext_vector_type(8)))  __bf16 v8bf;
typedef __attribute__((ext_vector_type(8)))  float  v8f;
typedef __attribute__((ext_vector_type(4)))  unsigned int v4u;
typedef __attribute__((ext_vector_type(8)))  unsigned int v8u;

union V16U { v16bf v; v8bf h[2]; };

__device__ __forceinline__ unsigned short f2bf(float f) {
  unsigned int u = __float_as_uint(f);
  u += 0x7FFFu + ((u >> 16) & 1u);      // round-to-nearest-even
  return (unsigned short)(u >> 16);
}

// A-fragment (16x32 bf16, MxK): lane<16 -> row=lane,   K [kb, kb+16)
//                               lane>=16 -> row=lane-16, K [kb+16, kb+32)
__device__ __forceinline__ v16bf load_a_frag(const unsigned short* row, int kb, int ka) {
  V16U u;
  u.h[0] = *(const v8bf*)(row + kb + 2 * ka);
  u.h[1] = *(const v8bf*)(row + kb + 2 * ka + 8);
  return u.v;
}

__device__ __forceinline__ v8f wmma_bf16(v16bf a, v16bf b, v8f c) {
  return __builtin_amdgcn_wmma_f32_16x16x32_bf16(false, a, false, b, (short)0, c,
                                                 false, false);
}

// ------------------------- kernel 1: RMSNorm + cast --------------------------
__global__ __launch_bounds__(256)
void k_rmsnorm(const float* __restrict__ x, const float* __restrict__ nw,
               float* __restrict__ xnf, unsigned short* __restrict__ xnb) {
  __shared__ float red[8];
  const int t    = blockIdx.x;
  const int lane = threadIdx.x & 31;
  const int wave = threadIdx.x >> 5;
  const float* xr = x + (size_t)t * DD;

  float v[8];
  float ss = 0.f;
#pragma unroll
  for (int i = 0; i < 8; ++i) {
    v[i] = xr[threadIdx.x + i * 256];
    ss += v[i] * v[i];
  }
#pragma unroll
  for (int off = 16; off; off >>= 1) ss += __shfl_xor(ss, off, 32);
  if (lane == 0) red[wave] = ss;
  __syncthreads();
  float tot = 0.f;
#pragma unroll
  for (int w = 0; w < 8; ++w) tot += red[w];
  const float r = rsqrtf(tot * (1.0f / (float)DD) + 1.1920929e-07f);

#pragma unroll
  for (int i = 0; i < 8; ++i) {
    const int c = threadIdx.x + i * 256;
    const float xv = v[i] * r * nw[c];
    xnf[(size_t)t * DD + c] = xv;
    xnb[(size_t)t * DD + c] = f2bf(xv);
  }
}

// ------------------------- kernel 2: f32 -> bf16 cast ------------------------
__global__ __launch_bounds__(256)
void k_cast_bf16(const float* __restrict__ src, unsigned short* __restrict__ dst, int n) {
  const int i = (blockIdx.x * 256 + threadIdx.x) * 4;
  if (i >= n) return;
  const float4 f = *(const float4*)(src + i);
  *(ushort4*)(dst + i) = make_ushort4(f2bf(f.x), f2bf(f.y), f2bf(f.z), f2bf(f.w));
}

// ---- kernel 3: router GEMM (WMMA) + top-32 + softmax + gene combine, fused ----
// One workgroup = 16 tokens. LDS: [0,64KB) bf16 A tile (16xD), [64KB,320KB) f32 logits (16xG).
// A tile is DMA'd in by the Tensor Data Mover (one descriptor per workgroup).
__global__ __launch_bounds__(256)
void k_router_topk_express(const unsigned short* __restrict__ xnb,
                           const unsigned short* __restrict__ Wcb,
                           const float* __restrict__ temperature,
                           const float* __restrict__ genes,
                           unsigned short* __restrict__ expr) {
  extern __shared__ unsigned char smem[];
  unsigned short* sA  = (unsigned short*)smem;                // 16 x DD bf16
  float*          sLg = (float*)(smem + (size_t)16 * DD * 2); // 16 x GG f32

  const int m0   = blockIdx.x * 16;
  const int lane = threadIdx.x & 31;
  const int wave = threadIdx.x >> 5;

  // ---- TDM: stage the 16 x 2048 bf16 activation tile into LDS --------------
  // D# group0: count=1 | lds_addr | 57-bit global addr | type=2 ("image").
  // D# group1: wg_mask=0 (not in cluster), data_size=3 (8B units),
  //            tensor_dim0=512 (units/row), dim0_stride=512, tensor_dim1=8192,
  //            tile_dim0=512, tile_dim1=16 (2D tile -> 2-group form).
  if (wave == 0) {
    const unsigned long long gaddr =
        (unsigned long long)(size_t)(const void*)(xnb + (size_t)m0 * DD);
    const unsigned int lds_off = (unsigned int)(size_t)(void*)sA;
    v4u g0d;
    g0d[0] = 1u;                                               // count=1, user desc
    g0d[1] = lds_off;                                          // lds_addr
    g0d[2] = (unsigned int)gaddr;                              // global_addr[31:0]
    g0d[3] = ((unsigned int)(gaddr >> 32) & 0x01FFFFFFu)       // global_addr[56:32]
             | (2u << 30);                                     // type=2
    v8u g1d;
    g1d[0] = (3u << 16);       // workgroup_mask=0, data_size=8B
    g1d[1] = (512u << 16);     // [63:48]  tensor_dim0[15:0] = 512
    g1d[2] = (8192u << 16);    // [95:80]  tensor_dim1[15:0] = 8192
    g1d[3] = (512u << 16);     // [127:112] tile_dim0 = 512
    g1d[4] = 16u;              // [143:128] tile_dim1 = 16
    g1d[5] = 512u;             // [191:160] tensor_dim0_stride[31:0] = 512
    g1d[6] = 0u;
    g1d[7] = 0u;
    asm volatile("tensor_load_to_lds %0, %1" :: "s"(g0d), "s"(g1d) : "memory");
    __builtin_amdgcn_s_wait_tensorcnt(0);
  }
  __syncthreads();

  const float invT = 1.0f / fmaxf(temperature[0], 0.1f);
  const unsigned short* Arow = sA + (size_t)(lane & 15) * DD;
  const int ka = (lane >> 4) * 8;

  // Each wave sweeps 16 pairs of N-tiles; the A fragment feeds 2 WMMAs.
  for (int tp = wave; tp < GG / 32; tp += 8) {
    const int g0 = tp * 32;
    const unsigned short* B0 =
        Wcb + (size_t)(g0 + (lane & 15)) * DD + (lane >> 4) * 16;
    const unsigned short* B1 = B0 + (size_t)16 * DD;
    v8f acc0 = {}, acc1 = {};
    for (int kb = 0; kb < DD; kb += 32) {
      __builtin_prefetch(B0 + kb + 256, 0, 3);
      __builtin_prefetch(B1 + kb + 256, 0, 3);
      const v16bf a = load_a_frag(Arow, kb, ka);     // ds_load_b128 x2, reused twice
      acc0 = wmma_bf16(a, *(const v16bf*)(B0 + kb), acc0);
      acc1 = wmma_bf16(a, *(const v16bf*)(B1 + kb), acc1);
    }
    const int mhi = 8 * (lane >> 4);
    const int n   = g0 + (lane & 15);
#pragma unroll
    for (int rrr = 0; rrr < 8; ++rrr) {
      sLg[(size_t)(mhi + rrr) * GG + n]      = acc0[rrr] * invT;
      sLg[(size_t)(mhi + rrr) * GG + n + 16] = acc1[rrr] * invT;
    }
  }
  __syncthreads();

  // Overlay top-k scratch on the (now dead) A-tile region.
  float* topv = (float*)(smem);
  int*   topi = (int*)(smem + 16 * KACT * 4);

  for (int tt = 0; tt < 2; ++tt) {
    const int t = wave * 2 + tt;       // each wave owns 2 tokens
    float* lg = sLg + (size_t)t * GG;

    // Iterative top-32: per-lane scan (stride-32, bank-conflict free) + shfl butterfly.
    for (int it = 0; it < KACT; ++it) {
      float best = -3.4e38f;
      int   bi   = lane;
      for (int j = lane; j < GG; j += 32) {
        const float v = lg[j];
        if (v > best) { best = v; bi = j; }
      }
#pragma unroll
      for (int off = 16; off; off >>= 1) {
        const float ov = __shfl_xor(best, off, 32);
        const int   oi = __shfl_xor(bi, off, 32);
        if (ov > best || (ov == best && oi < bi)) { best = ov; bi = oi; }
      }
      if (lane == 0) {
        topv[t * KACT + it] = best;
        topi[t * KACT + it] = bi;
        lg[bi] = -3.4e38f;             // knock out for next iteration (in-wave DS is in-order)
      }
    }

    // Softmax over the 32 selected values (descending -> index 0 is the max).
    const float e = __expf(topv[t * KACT + lane] - topv[t * KACT]);
    float s = e;
#pragma unroll
    for (int off = 16; off; off >>= 1) s += __shfl_xor(s, off, 32);
    const float wk_mine = e / s;

    // expressed[t, :] = sum_k w_k * genes[idx_k, :]; lane owns 8 contiguous channels.
    float acc[8] = {0.f, 0.f, 0.f, 0.f, 0.f, 0.f, 0.f, 0.f};
    const int c0 = lane * 8;
    for (int k = 0; k < KACT; ++k) {
      const float wk = __shfl(wk_mine, k, 32);
      const float* gr = genes + (size_t)topi[t * KACT + k] * GD + c0;
      const float4 ga = *(const float4*)(gr);
      const float4 gb = *(const float4*)(gr + 4);
      acc[0] += wk * ga.x; acc[1] += wk * ga.y;
      acc[2] += wk * ga.z; acc[3] += wk * ga.w;
      acc[4] += wk * gb.x; acc[5] += wk * gb.y;
      acc[6] += wk * gb.z; acc[7] += wk * gb.w;
    }
    unsigned short* eo = expr + (size_t)(m0 + t) * GD + c0;
    *(ushort4*)(eo)     = make_ushort4(f2bf(acc[0]), f2bf(acc[1]), f2bf(acc[2]), f2bf(acc[3]));
    *(ushort4*)(eo + 4) = make_ushort4(f2bf(acc[4]), f2bf(acc[5]), f2bf(acc[6]), f2bf(acc[7]));
  }
}

// ------- kernel 4: down/up WMMA GEMMs + tanh/mul epilogue, K=256 unrolled -------
__global__ __launch_bounds__(256)
void k_downup(const unsigned short* __restrict__ expr,
              const unsigned short* __restrict__ Wdb,
              const unsigned short* __restrict__ Wub,
              const float* __restrict__ xnf,
              const float* __restrict__ scale,
              float* __restrict__ out) {
  const int lane = threadIdx.x & 31;
  const int wave = threadIdx.x >> 5;
  const int m0 = blockIdx.x * 16;
  const int n0 = blockIdx.y * 128 + wave * 16;

  const unsigned short* Arow = expr + (size_t)(m0 + (lane & 15)) * GD;
  const int ka = (lane >> 4) * 8;
  const unsigned short* Bd = Wdb + (size_t)(n0 + (lane & 15)) * GD + (lane >> 4) * 16;
  const unsigned short* Bu = Wub + (size_t)(n0 + (lane & 15)) * GD + (lane >> 4) * 16;

  v8f ad = {}, au = {};
#pragma unroll
  for (int kb = 0; kb < GD; kb += 32) {
    const v16bf a = load_a_frag(Arow, kb, ka);   // shared between both GEMMs
    ad = wmma_bf16(a, *(const v16bf*)(Bd + kb), ad);
    au = wmma_bf16(a, *(const v16bf*)(Bu + kb), au);
  }

  const float sc = scale[0];
  const int mh = m0 + 8 * (lane >> 4);
  const int n  = n0 + (lane & 15);
#pragma unroll
  for (int rrr = 0; rrr < 8; ++rrr) {
    const size_t idx = (size_t)(mh + rrr) * DD + n;
    out[idx] = tanhf(ad[rrr]) * xnf[idx] * au[rrr] * sc;
  }
}

// --------------------------------- launcher ----------------------------------
extern "C" void kernel_launch(void* const* d_in, const int* in_sizes, int n_in,
                              void* d_out, int out_size, void* d_ws, size_t ws_size,
                              hipStream_t stream) {
  const float* x     = (const float*)d_in[0];   // (B,T,D)
  const float* Wc    = (const float*)d_in[1];   // (G,D)
  const float* temp  = (const float*)d_in[2];   // scalar
  const float* genes = (const float*)d_in[3];   // (G,GD)
  const float* Wd    = (const float*)d_in[4];   // (D,GD)
  const float* Wu    = (const float*)d_in[5];   // (D,GD)
  const float* nw    = (const float*)d_in[6];   // (D,)
  const float* scale = (const float*)d_in[7];   // scalar
  float* out = (float*)d_out;

  unsigned char* ws = (unsigned char*)d_ws;
  size_t o = 0;
  float*          xnf  = (float*)(ws + o);          o += (size_t)BT * DD * 4;
  unsigned short* xnb  = (unsigned short*)(ws + o); o += (size_t)BT * DD * 2;
  unsigned short* Wcb  = (unsigned short*)(ws + o); o += (size_t)GG * DD * 2;
  unsigned short* Wdb  = (unsigned short*)(ws + o); o += (size_t)DD * GD * 2;
  unsigned short* Wub  = (unsigned short*)(ws + o); o += (size_t)DD * GD * 2;
  unsigned short* expr = (unsigned short*)(ws + o); o += (size_t)BT * GD * 2;
  (void)o; (void)ws_size; (void)in_sizes; (void)n_in; (void)out_size;

  k_rmsnorm<<<BT, 256, 0, stream>>>(x, nw, xnf, xnb);
  k_cast_bf16<<<(GG * DD / 4) / 256, 256, 0, stream>>>(Wc, Wcb, GG * DD);
  k_cast_bf16<<<(DD * GD / 4) / 256, 256, 0, stream>>>(Wd, Wdb, DD * GD);
  k_cast_bf16<<<(DD * GD / 4) / 256, 256, 0, stream>>>(Wu, Wub, DD * GD);

  const size_t shmem = (size_t)16 * DD * 2 + (size_t)16 * GG * 4;  // 64KB + 256KB = 320KB
  (void)hipFuncSetAttribute((const void*)k_router_topk_express,
                            hipFuncAttributeMaxDynamicSharedMemorySize, (int)shmem);
  k_router_topk_express<<<BT / 16, 256, shmem, stream>>>(xnb, Wcb, temp, genes, expr);

  k_downup<<<dim3(BT / 16, DD / 128), 256, 0, stream>>>(expr, Wdb, Wub, xnf, scale, out);
}
